// OriAttention_10934986736340
// MI455X (gfx1250) — matrix operand order repaired and checked by reading, not compiled, not run on hardware
//
#include <hip/hip_runtime.h>

// ---------------------------------------------------------------------------
// OriAttention (B=8, L=4096, D=64) for gfx1250.
// scores and pair-mask are symmetric, so the reference's softmax-over-q is
// standard attention with Q=K=V=x, scale = 1/64^0.25, masked fill = -1e7
// (finite fill -> fully-masked rows softmax to uniform; this flash loop
// reproduces that since the running max saturates at -1e7).
// ---------------------------------------------------------------------------

typedef __attribute__((ext_vector_type(16))) __bf16          v16bf;
typedef __attribute__((ext_vector_type(2)))  __bf16          v2bf;
typedef __attribute__((ext_vector_type(8)))  float           v8f;
typedef __attribute__((ext_vector_type(8)))  unsigned short  u16x8;
typedef __attribute__((ext_vector_type(4)))  unsigned int    u32x4;
typedef __attribute__((ext_vector_type(4)))  float           f32x4;

#define B_        8
#define L_        4096
#define D_        64
#define KBLK      128                 // keys staged in LDS per block iter
#define WAVES     8
#define NTHREADS  (WAVES * 32)
#define ROWS_PB   (WAVES * 16)        // 128 query rows per block

__device__ __forceinline__ unsigned short f2bf(float f) {
  unsigned int u = __builtin_bit_cast(unsigned int, f);
  u += 0x7fffu + ((u >> 16) & 1u);     // round-to-nearest-even
  return (unsigned short)(u >> 16);
}

// pack two floats -> two bf16 in one dword (lo -> bits[15:0])
__device__ __forceinline__ unsigned int f2bf2(float lo, float hi) {
#if __has_builtin(__builtin_amdgcn_cvt_pk_bf16_f32)
  v2bf t = __builtin_amdgcn_cvt_pk_bf16_f32(lo, hi);   // v_cvt_pk_bf16_f32
  return __builtin_bit_cast(unsigned int, t);
#else
  return ((unsigned int)f2bf(hi) << 16) | (unsigned int)f2bf(lo);
#endif
}

union BF16Frag {
  v16bf          v;
  unsigned short u[16];
  unsigned int   w[8];
  u16x8          h[2];
};

union PkBuf { unsigned int w[16]; u32x4 q[4]; };

__global__ __launch_bounds__(NTHREADS)
void ori_attn_kernel(const float* __restrict__ x,
                     const int*   __restrict__ mask,
                     float*       __restrict__ out) {
  // LDS staging (16B-aligned rows for b128 ds loads)
  __shared__ __align__(16) unsigned short kt [KBLK][D_ + 8];    // [key][d]
  __shared__ __align__(16) unsigned short ktT[D_][KBLK + 8];    // [d][key]
  __shared__ __align__(16) unsigned short pscr[WAVES][16][32];  // P tiles
  __shared__ int km[KBLK];

  const int tid    = threadIdx.x;
  const int wave   = tid >> 5;
  const int lane   = tid & 31;
  const int lane16 = lane & 15;
  const int half   = lane >> 4;

  const int blocksPerBatch = L_ / ROWS_PB;
  const int b     = blockIdx.x / blocksPerBatch;
  const int qblk  = (blockIdx.x % blocksPerBatch) * ROWS_PB;
  const int qbase = qblk + wave * 16;

  const float SCALE = 0.35355339059327373f;   // 1 / 64^0.25 (folded into Q)

  // ---- Q fragments (A operands): 16 rows x 64 d, bf16, pre-scaled ----
  // A k order used here (16x32, B matches): lane<16 holds row=lane, K={0..15};
  //                   lane>=16 holds row=lane-16, K={16..31}.
  BF16Frag qa[2];
  {
    const size_t xrow = ((size_t)b * L_ + (size_t)(qbase + lane16)) * D_;
#pragma unroll
    for (int c = 0; c < 2; ++c) {
      const float* p0 = x + xrow + 32 * c + half * 16;       // K-chunk low
      const float* p1 = x + xrow + 32 * c + half * 16 + 8;   // K-chunk high
#pragma unroll
      for (int q = 0; q < 2; ++q) {
        f32x4 a  = *(const f32x4*)(p0 + 4 * q);
        f32x4 bb = *(const f32x4*)(p1 + 4 * q);
        qa[c].w[2 * q]     = f2bf2(a[0] * SCALE, a[1] * SCALE);
        qa[c].w[2 * q + 1] = f2bf2(a[2] * SCALE, a[3] * SCALE);
        qa[c].w[4 + 2 * q]     = f2bf2(bb[0] * SCALE, bb[1] * SCALE);
        qa[c].w[4 + 2 * q + 1] = f2bf2(bb[2] * SCALE, bb[3] * SCALE);
      }
    }
  }

  // query-token mask, one value per owned row (row m = r + 8*half)
  int qm[8];
#pragma unroll
  for (int r = 0; r < 8; ++r)
    qm[r] = mask[(size_t)b * L_ + qbase + half * 8 + r];

  // online-softmax state (per owned row) + output accumulators (C layout)
  float mrow[8], lrow[8];
  v8f o[4];
#pragma unroll
  for (int r = 0; r < 8; ++r) { mrow[r] = -3.0e38f; lrow[r] = 0.0f; }
#pragma unroll
  for (int g = 0; g < 4; ++g) o[g] = (v8f){0,0,0,0,0,0,0,0};

  for (int kb = 0; kb < L_; kb += KBLK) {
    // speculative prefetch of the next key block
    if (kb + KBLK < L_)
      __builtin_prefetch(x + ((size_t)b * L_ + kb + KBLK) * D_ + tid * 32, 0, 1);

    // ---- stage key-major tile: one thread = one (key, 32-d half) ----
    {
      const int key = tid >> 1;
      const int dp  = (tid & 1) * 32;
      const f32x4* src = (const f32x4*)(x + ((size_t)b * L_ + kb + key) * D_ + dp);
      PkBuf pk;
#pragma unroll
      for (int q = 0; q < 8; ++q) {
        f32x4 v = src[q];
        pk.w[2 * q]     = f2bf2(v[0], v[1]);
        pk.w[2 * q + 1] = f2bf2(v[2], v[3]);
      }
#pragma unroll
      for (int s = 0; s < 4; ++s)
        *(u32x4*)&kt[key][dp + 8 * s] = pk.q[s];
    }
    // ---- stage d-major tile: one thread = one (d, key-pair), packed b32 ----
#pragma unroll
    for (int it = 0; it < (D_ * KBLK / 2) / NTHREADS; ++it) {   // 16 iters
      const int idx = tid + it * NTHREADS;
      const int d   = idx & (D_ - 1);
      const int kp  = idx >> 6;                                  // key pair
      const float* base = x + ((size_t)b * L_ + kb + 2 * kp) * D_ + d;
      *(unsigned int*)&ktT[d][2 * kp] = f2bf2(base[0], base[D_]);
    }
    if (tid < KBLK) km[tid] = mask[(size_t)b * L_ + kb + tid];
    __syncthreads();

    // ---- consume 32 keys per step: 4 WMMAs (S) + 4 WMMAs (PV) ----
    for (int cp = 0; cp < KBLK; cp += 32) {
      const int key0 = cp + lane16;        // column tile 0
      const int key1 = cp + 16 + lane16;   // column tile 1
      const int db   = half * 16;          // B layout K-half select

      // batch all 8 B-fragment loads before the WMMAs (one dscnt wait)
      BF16Frag b00, b01, b10, b11;
      b00.h[0] = *(const u16x8*)&kt[key0][db];
      b00.h[1] = *(const u16x8*)&kt[key0][db + 8];
      b01.h[0] = *(const u16x8*)&kt[key0][32 + db];
      b01.h[1] = *(const u16x8*)&kt[key0][32 + db + 8];
      b10.h[0] = *(const u16x8*)&kt[key1][db];
      b10.h[1] = *(const u16x8*)&kt[key1][db + 8];
      b11.h[0] = *(const u16x8*)&kt[key1][32 + db];
      b11.h[1] = *(const u16x8*)&kt[key1][32 + db + 8];

      v8f zero = (v8f){0,0,0,0,0,0,0,0};
      v8f s0 = __builtin_amdgcn_wmma_f32_16x16x32_bf16(false, qa[0].v, false, b00.v,
                                                       (short)0, zero, false, false);
      v8f s1 = __builtin_amdgcn_wmma_f32_16x16x32_bf16(false, qa[0].v, false, b10.v,
                                                       (short)0, zero, false, false);
      s0 = __builtin_amdgcn_wmma_f32_16x16x32_bf16(false, qa[1].v, false, b01.v,
                                                   (short)0, s0, false, false);
      s1 = __builtin_amdgcn_wmma_f32_16x16x32_bf16(false, qa[1].v, false, b11.v,
                                                   (short)0, s1, false, false);

      // ---- mask (C layout: row = r + 8*half, col = lane16) ----
      const int k0 = km[cp + lane16];
      const int k1 = km[cp + 16 + lane16];
#pragma unroll
      for (int r = 0; r < 8; ++r) {
        s0[r] = (qm[r] && k0) ? s0[r] : -1.0e7f;
        s1[r] = (qm[r] && k1) ? s1[r] : -1.0e7f;
      }

      // ---- online softmax; write P (bf16) to per-wave LDS scratch ----
      float corr[8];
#pragma unroll
      for (int r = 0; r < 8; ++r) {
        float t = fmaxf(s0[r], s1[r]);
#pragma unroll
        for (int mm = 1; mm < 16; mm <<= 1)      // stays within half-wave
          t = fmaxf(t, __shfl_xor(t, mm, 32));
        float mn = fmaxf(mrow[r], t);
        corr[r]  = __expf(mrow[r] - mn);
        mrow[r]  = mn;
        float p0 = __expf(s0[r] - mn);
        float p1 = __expf(s1[r] - mn);
        float rs = p0 + p1;
#pragma unroll
        for (int mm = 1; mm < 16; mm <<= 1)
          rs += __shfl_xor(rs, mm, 32);
        lrow[r] = lrow[r] * corr[r] + rs;
        const int m = r + half * 8;
        pscr[wave][m][lane16]      = f2bf(p0);
        pscr[wave][m][16 + lane16] = f2bf(p1);
      }
#pragma unroll
      for (int g = 0; g < 4; ++g)
#pragma unroll
        for (int r = 0; r < 8; ++r)
          o[g][r] *= corr[r];

      asm volatile("s_wait_dscnt 0" ::: "memory");   // P stores -> P reload

      // P as A operand (16x32)
      BF16Frag pa;
      pa.h[0] = *(const u16x8*)&pscr[wave][lane16][half * 16];
      pa.h[1] = *(const u16x8*)&pscr[wave][lane16][half * 16 + 8];

      // ---- O += P * V : batch the 4 V fragments, then 4 independent WMMAs
      BF16Frag vb[4];
      const int kbase = cp + half * 16;
#pragma unroll
      for (int g = 0; g < 4; ++g) {
        const int dcol = g * 16 + lane16;
        vb[g].h[0] = *(const u16x8*)&ktT[dcol][kbase];
        vb[g].h[1] = *(const u16x8*)&ktT[dcol][kbase + 8];
      }
#pragma unroll
      for (int g = 0; g < 4; ++g)
        o[g] = __builtin_amdgcn_wmma_f32_16x16x32_bf16(false, pa.v, false, vb[g].v,
                                                       (short)0, o[g], false, false);
    }
    __syncthreads();
  }

  // ---- epilogue: normalize and store (C layout scatter) ----
#pragma unroll
  for (int r = 0; r < 8; ++r) {
    const float inv = 1.0f / lrow[r];
    float* orow = out + ((size_t)b * L_ + (qbase + half * 8 + r)) * D_;
#pragma unroll
    for (int g = 0; g < 4; ++g)
      orow[g * 16 + lane16] = o[g][r] * inv;
  }
}

extern "C" void kernel_launch(void* const* d_in, const int* in_sizes, int n_in,
                              void* d_out, int out_size, void* d_ws, size_t ws_size,
                              hipStream_t stream) {
  (void)in_sizes; (void)n_in; (void)out_size; (void)d_ws; (void)ws_size;
  const float* x    = (const float*)d_in[0];
  const int*   mask = (const int*)d_in[1];
  float*       out  = (float*)d_out;

  dim3 grid(B_ * (L_ / ROWS_PB));   // 8 * 32 = 256 blocks
  dim3 block(NTHREADS);             // 256 threads = 8 wave32
  ori_attn_kernel<<<grid, block, 0, stream>>>(x, mask, out);
}
